// ChannelAttentionBlock_30803505447432
// MI455X (gfx1250) — matrix operand order repaired and checked by reading, not compiled, not run on hardware
//
#include <hip/hip_runtime.h>
#include <hip/hip_bf16.h>
#include <math.h>

// Problem constants (match reference)
#define BB    8
#define CCH   256
#define SCH   64
#define HH    128
#define WW    128
#define HWC   (HH*WW)
#define NHEAD 8

typedef __attribute__((ext_vector_type(16))) __bf16 v16bf;
typedef __attribute__((ext_vector_type(8)))  __bf16 v8bf;
typedef __attribute__((ext_vector_type(8)))  float  v8f;
typedef __attribute__((ext_vector_type(4)))  unsigned v4u;

static __device__ __forceinline__ unsigned pack_bf16(float a, float b) {
  unsigned short ua = __builtin_bit_cast(unsigned short, (__bf16)a);
  unsigned short ub = __builtin_bit_cast(unsigned short, (__bf16)b);
  return (unsigned)ua | ((unsigned)ub << 16);   // lowers to v_cvt_pk_bf16_f32
}

// ---------------------------------------------------------------------------
// Implicit-GEMM conv (1x1 or 3x3, zero pad=ksize/2) via WMMA bf16, f32 accum.
// Tile: M=64 (Cout block) x N=256 pixels, 8 waves; wave w owns pixels
// [pBase+32w, +32) for all 64 output channels -> 8 WMMA accumulators, with
// A fragments reused across both N sub-blocks (12 ds_load_b128 per 8 WMMA).
// Double-buffered LDS, one barrier per K-step; next step's global loads are
// issued before the WMMAs to hide HBM latency. Incremental (dy,dx) tap state
// avoids per-step integer division. Optional fused channel-LayerNorm epilogue
// (requires cout==64, gridDim.y==1). Channel-concat via (in0,cin0)+(in1,...).
// ---------------------------------------------------------------------------
__launch_bounds__(256)
__global__ void wmma_conv(const float* __restrict__ in0, const float* __restrict__ in1,
                          int cin0, int cin,
                          const float* __restrict__ wgt,   // [cout, cin, k, k]
                          float* __restrict__ out, int cout, int ksize,
                          const float* __restrict__ lng, const float* __restrict__ lnb,
                          int fuse_ln)
{
  const int tid  = threadIdx.x;
  const int lane = tid & 31;
  const int wave = tid >> 5;
  const int l15  = lane & 15;
  const int hi   = lane >> 4;           // 0: lanes 0-15, 1: lanes 16-31
  const int pBase  = blockIdx.x * 256;
  const int ocBase = blockIdx.y * 64;
  const int b      = blockIdx.z;
  const int ktaps  = ksize * ksize;

  // Per-thread staging coordinates (fixed for whole kernel)
  const int nB  = tid;                  // B: this thread's pixel (stages all 32 ch)
  const int kpA = tid & 15;             // A: channel-pair (k = 2*kpA, 2*kpA+1)
  const int mA  = tid >> 4;             // A: starting output channel row

  __shared__ __attribute__((aligned(32))) __bf16 As[2][64 * 32];   // [m][k]
  __shared__ __attribute__((aligned(32))) __bf16 Bs[2][256 * 40];  // [n][k] pad40
  __shared__ float gls[64], bls[64];

  if (fuse_ln && tid < 64) { gls[tid] = lng[tid]; bls[tid] = lnb[tid]; }

  v8f accA0 = {0,0,0,0,0,0,0,0}, accA1 = {0,0,0,0,0,0,0,0};
  v8f accA2 = {0,0,0,0,0,0,0,0}, accA3 = {0,0,0,0,0,0,0,0};
  v8f accB0 = {0,0,0,0,0,0,0,0}, accB1 = {0,0,0,0,0,0,0,0};
  v8f accB2 = {0,0,0,0,0,0,0,0}, accB3 = {0,0,0,0,0,0,0,0};

  // Fragment K bases: contiguous k = 16*hi + s, same order for A and B
  const int kbA   = hi ? 16 : 0;        // A: lanes16-31 hold K 16-31
  const int kbB   = hi ? 16 : 0;        // B: lanes16-31 hold K 16-31
  const int nloc0 = wave * 32 + l15;    // N sub-block 0
  const int nloc1 = nloc0 + 16;         // N sub-block 1

  unsigned apack[4], bpack[16];

  // ---- load one K-step's A/B tiles into registers (bf16-pair packed) ----
  auto load_tile = [&](int t, int icb, int dy, int dx) {
    // B: 32 channels of this thread's pixel
    {
      const int p  = pBase + nB;
      const int py = (p >> 7) + dy;          // W == 128
      const int px = (p & 127) + dx;
      const bool valid = (py >= 0 && py < HH && px >= 0 && px < WW);
      const int off = valid ? (py * WW + px) : 0;
      const float* sbase = (icb < cin0)      // scalar select (chunk-uniform)
          ? in0 + ((size_t)b * cin0 + icb) * HWC
          : in1 + ((size_t)b * (cin - cin0) + (icb - cin0)) * HWC;
      const float* pth = sbase + off;
      #pragma unroll
      for (int i = 0; i < 16; ++i) {
        float f0 = pth[0];
        float f1 = pth[HWC];
        if (!valid) { f0 = 0.f; f1 = 0.f; }
        bpack[i] = pack_bf16(f0, f1);
        pth += 2 * HWC;
      }
    }
    // A: 64x32 weights, this thread: rows mA+16i, chans 2*kpA, 2*kpA+1
    {
      const float* wp = wgt + ((size_t)(ocBase + mA) * cin + icb + 2 * kpA) * ktaps + t;
      const size_t stride = (size_t)16 * cin * ktaps;
      #pragma unroll
      for (int i = 0; i < 4; ++i) {
        apack[i] = pack_bf16(wp[0], wp[ktaps]);
        wp += stride;
      }
    }
  };

  auto store_tile = [&](int par) {
    unsigned* Au = (unsigned*)As[par];
    unsigned* Bu = (unsigned*)Bs[par];
    #pragma unroll
    for (int i = 0; i < 4; ++i) Au[(mA + 16 * i) * 16 + kpA] = apack[i];
    #pragma unroll
    for (int j = 0; j < 4; ++j) {
      v4u w = {bpack[4*j], bpack[4*j+1], bpack[4*j+2], bpack[4*j+3]};
      *(v4u*)&Bu[nB * 20 + 4 * j] = w;        // 80B row stride, 16B aligned
    }
  };

  const int nsteps = ktaps * (cin >> 5);
  const int kmax = (ksize == 3) ? 1 : 0;
  int t = 0, icb = 0;
  int dyc = -kmax, dxc = -kmax;
  load_tile(0, 0, dyc, dxc);

  for (int step = 0; step < nsteps; ++step) {
    const int par = step & 1;
    store_tile(par);
    __syncthreads();
    if (step + 1 < nsteps) {            // issue next step's loads early
      icb += 32;
      if (icb == cin) {                 // tap advance, no division
        icb = 0; ++t;
        ++dxc; if (dxc > kmax) { dxc = -kmax; ++dyc; }
      }
      load_tile(t, icb, dyc, dxc);
    }
    const __bf16* Ab = As[par];
    const __bf16* Bb = Bs[par];
    v8bf b00 = *(const v8bf*)&Bb[nloc0 * 40 + kbB];
    v8bf b01 = *(const v8bf*)&Bb[nloc0 * 40 + kbB + 8];
    v16bf bf0 = __builtin_shufflevector(b00, b01,
        0,1,2,3,4,5,6,7,8,9,10,11,12,13,14,15);
    v8bf b10 = *(const v8bf*)&Bb[nloc1 * 40 + kbB];
    v8bf b11 = *(const v8bf*)&Bb[nloc1 * 40 + kbB + 8];
    v16bf bf1 = __builtin_shufflevector(b10, b11,
        0,1,2,3,4,5,6,7,8,9,10,11,12,13,14,15);
#define AFRAG(F, ACCA, ACCB)                                                   \
    {                                                                          \
      v8bf a0 = *(const v8bf*)&Ab[((F)*16 + l15) * 32 + kbA];                  \
      v8bf a1 = *(const v8bf*)&Ab[((F)*16 + l15) * 32 + kbA + 8];              \
      v16bf af = __builtin_shufflevector(a0, a1,                               \
          0,1,2,3,4,5,6,7,8,9,10,11,12,13,14,15);                              \
      ACCA = __builtin_amdgcn_wmma_f32_16x16x32_bf16(                          \
          false, af, false, bf0, (short)0, ACCA, false, false);                \
      ACCB = __builtin_amdgcn_wmma_f32_16x16x32_bf16(                          \
          false, af, false, bf1, (short)0, ACCB, false, false);                \
    }
    AFRAG(0, accA0, accB0)
    AFRAG(1, accA1, accB1)
    AFRAG(2, accA2, accB2)
    AFRAG(3, accA3, accB3)
#undef AFRAG
  }

  // ---- optional fused channel LayerNorm (cout==64: block owns all chans) ----
  if (fuse_ln) {
    const int mb = hi * 8;
#define LNBLK(A0, A1, A2, A3)                                                  \
    {                                                                          \
      float s = 0.f, s2 = 0.f;                                                 \
      _Pragma("unroll")                                                        \
      for (int r = 0; r < 8; ++r) {                                            \
        s  += A0[r] + A1[r] + A2[r] + A3[r];                                   \
        s2 += A0[r]*A0[r] + A1[r]*A1[r] + A2[r]*A2[r] + A3[r]*A3[r];           \
      }                                                                        \
      s  += __shfl_xor(s, 16);          /* partner half-lane: other 32 ch */   \
      s2 += __shfl_xor(s2, 16);                                                \
      float mu = s * 0.015625f;                                                \
      float rs = rsqrtf(s2 * 0.015625f - mu * mu + 1e-5f);                     \
      _Pragma("unroll")                                                        \
      for (int r = 0; r < 8; ++r) {                                            \
        A0[r] = (A0[r] - mu) * rs * gls[mb +      r] + bls[mb +      r];       \
        A1[r] = (A1[r] - mu) * rs * gls[mb + 16 + r] + bls[mb + 16 + r];       \
        A2[r] = (A2[r] - mu) * rs * gls[mb + 32 + r] + bls[mb + 32 + r];       \
        A3[r] = (A3[r] - mu) * rs * gls[mb + 48 + r] + bls[mb + 48 + r];       \
      }                                                                        \
    }
    LNBLK(accA0, accA1, accA2, accA3)
    LNBLK(accB0, accB1, accB2, accB3)
#undef LNBLK
  }

  // ---- epilogue: C layout lanes0-15 -> M=r, lanes16-31 -> M=r+8 ----
  size_t ob = ((size_t)b * cout + ocBase) * HWC + pBase;
  #pragma unroll
  for (int r = 0; r < 8; ++r) {
    int m0 = (hi ? 8 : 0) + r;
    out[ob + (size_t)(m0 +  0) * HWC + nloc0] = accA0[r];
    out[ob + (size_t)(m0 + 16) * HWC + nloc0] = accA1[r];
    out[ob + (size_t)(m0 + 32) * HWC + nloc0] = accA2[r];
    out[ob + (size_t)(m0 + 48) * HWC + nloc0] = accA3[r];
    out[ob + (size_t)(m0 +  0) * HWC + nloc1] = accB0[r];
    out[ob + (size_t)(m0 + 16) * HWC + nloc1] = accB1[r];
    out[ob + (size_t)(m0 + 32) * HWC + nloc1] = accB2[r];
    out[ob + (size_t)(m0 + 48) * HWC + nloc1] = accB3[r];
  }
}

// ---------------------------------------------------------------------------
// In-place channel LayerNorm (C=256 path). grid: (HWC/256, B)
// ---------------------------------------------------------------------------
__launch_bounds__(256)
__global__ void cln_kernel(float* __restrict__ x, const float* __restrict__ g,
                           const float* __restrict__ bt, int Cc)
{
  int p = blockIdx.x * 256 + threadIdx.x;
  int b = blockIdx.y;
  float* base = x + (size_t)b * Cc * HWC + p;
  float s = 0.f, s2 = 0.f;
  for (int c = 0; c < Cc; ++c) {
    float v = base[(size_t)c * HWC];
    s += v; s2 += v * v;
  }
  float mu  = s / (float)Cc;
  float var = s2 / (float)Cc - mu * mu;
  float rs  = rsqrtf(var + 1e-5f);
  for (int c = 0; c < Cc; ++c) {
    float v = base[(size_t)c * HWC];
    base[(size_t)c * HWC] = (v - mu) * rs * g[c] + bt[c];
  }
}

// ---------------------------------------------------------------------------
// Depthwise 3x3 conv (pad 1), optional exact GELU. grid: (HWC/256, Cc, B)
// ---------------------------------------------------------------------------
__launch_bounds__(256)
__global__ void dw_conv(const float* __restrict__ in, const float* __restrict__ w,
                        float* __restrict__ out, int dogelu)
{
  int p = blockIdx.x * 256 + threadIdx.x;
  int c = blockIdx.y, b = blockIdx.z, Cc = gridDim.y;
  int py = p >> 7, px = p & 127;        // W==128
  const float* src = in + ((size_t)b * Cc + c) * HWC;
  float acc = 0.f;
  #pragma unroll
  for (int t = 0; t < 9; ++t) {
    int dy = t / 3 - 1, dx = t % 3 - 1;
    int yy = py + dy, xx = px + dx;
    if (yy >= 0 && yy < HH && xx >= 0 && xx < WW)
      acc += w[c * 9 + t] * src[yy * WW + xx];
  }
  if (dogelu) acc = 0.5f * acc * (1.f + erff(acc * 0.70710678118654752f));
  out[((size_t)b * Cc + c) * HWC + p] = acc;
}

// ---------------------------------------------------------------------------
// Inverse L2 row-norms of q (and of k inside kv buffer). grid: (B*S, 2)
// ---------------------------------------------------------------------------
__launch_bounds__(256)
__global__ void rownorm_kernel(const float* __restrict__ q,
                               const float* __restrict__ kvbuf,
                               float* __restrict__ rnq, float* __restrict__ rnk)
{
  __shared__ float red[256];
  int r = blockIdx.x;                 // 0..B*S-1
  int isk = blockIdx.y;
  int b = r / SCH, c = r % SCH;
  const float* src = isk ? kvbuf + ((size_t)b * 2 * SCH + c) * HWC
                         : q     + ((size_t)b * SCH + c) * HWC;
  float s = 0.f;
  for (int n = threadIdx.x; n < HWC; n += 256) { float v = src[n]; s += v * v; }
  red[threadIdx.x] = s;
  __syncthreads();
  for (int st = 128; st > 0; st >>= 1) {
    if (threadIdx.x < st) red[threadIdx.x] += red[threadIdx.x + st];
    __syncthreads();
  }
  if (threadIdx.x == 0) {
    float inv = 1.f / fmaxf(sqrtf(red[0]), 1e-12f);
    (isk ? rnk : rnq)[r] = inv;
  }
}

// ---------------------------------------------------------------------------
// attn[b,h] = softmax_j( (q_i . k_j) * rnq_i * rnk_j * temp[h] ), 8x8 per head.
// ---------------------------------------------------------------------------
__launch_bounds__(256)
__global__ void attn_mat(const float* __restrict__ q, const float* __restrict__ kvbuf,
                         const float* __restrict__ rnq, const float* __restrict__ rnk,
                         const float* __restrict__ temp, float* __restrict__ attn)
{
  __shared__ float Gs[64];
  __shared__ float Gr[64];
  int bh = blockIdx.x, b = bh >> 3, h = bh & 7;
  const float* qb = q     + ((size_t)b * SCH + h * 8) * HWC;
  const float* kb = kvbuf + ((size_t)b * 2 * SCH + h * 8) * HWC;
  float acc[64];
  #pragma unroll
  for (int e = 0; e < 64; ++e) acc[e] = 0.f;
  for (int n = threadIdx.x; n < HWC; n += 256) {
    float qv[8], kv[8];
    #pragma unroll
    for (int i = 0; i < 8; ++i) {
      qv[i] = qb[(size_t)i * HWC + n];
      kv[i] = kb[(size_t)i * HWC + n];
    }
    #pragma unroll
    for (int i = 0; i < 8; ++i)
      #pragma unroll
      for (int j = 0; j < 8; ++j)
        acc[i * 8 + j] += qv[i] * kv[j];
  }
  if (threadIdx.x < 64) Gs[threadIdx.x] = 0.f;
  __syncthreads();
  #pragma unroll
  for (int e = 0; e < 64; ++e) atomicAdd(&Gs[e], acc[e]);
  __syncthreads();
  if (threadIdx.x < 64) {
    int i = threadIdx.x >> 3, j = threadIdx.x & 7;
    Gr[threadIdx.x] = Gs[threadIdx.x] * rnq[b * SCH + h * 8 + i]
                                      * rnk[b * SCH + h * 8 + j] * temp[h];
  }
  __syncthreads();
  if (threadIdx.x < 64) {
    int i = threadIdx.x >> 3;
    float mx = -1e30f;
    for (int jj = 0; jj < 8; ++jj) mx = fmaxf(mx, Gr[i * 8 + jj]);
    float sum = 0.f;
    for (int jj = 0; jj < 8; ++jj) sum += __expf(Gr[i * 8 + jj] - mx);
    attn[bh * 64 + threadIdx.x] = __expf(Gr[threadIdx.x] - mx) / sum;
  }
}

// ---------------------------------------------------------------------------
// o[b, h*8+i, p] = sum_d attn[i,d] * v[b, h*8+d, p].  grid: (HWC/256, B*NHEAD)
// ---------------------------------------------------------------------------
__launch_bounds__(256)
__global__ void attn_apply(const float* __restrict__ attn,
                           const float* __restrict__ kvbuf,
                           float* __restrict__ o)
{
  __shared__ float a[64];
  int bh = blockIdx.y, b = bh >> 3, h = bh & 7;
  if (threadIdx.x < 64) a[threadIdx.x] = attn[bh * 64 + threadIdx.x];
  __syncthreads();
  int p = blockIdx.x * 256 + threadIdx.x;
  const float* vb = kvbuf + ((size_t)b * 2 * SCH + SCH + h * 8) * HWC + p;
  float* ob = o + ((size_t)b * SCH + h * 8) * HWC + p;
  float vv[8];
  #pragma unroll
  for (int d = 0; d < 8; ++d) vv[d] = vb[(size_t)d * HWC];
  #pragma unroll
  for (int i = 0; i < 8; ++i) {
    float s = 0.f;
    #pragma unroll
    for (int d = 0; d < 8; ++d) s += a[i * 8 + d] * vv[d];
    ob[(size_t)i * HWC] = s;
  }
}

// ---------------------------------------------------------------------------
extern "C" void kernel_launch(void* const* d_in, const int* in_sizes, int n_in,
                              void* d_out, int out_size, void* d_ws, size_t ws_size,
                              hipStream_t stream)
{
  (void)in_sizes; (void)n_in; (void)out_size; (void)ws_size;
  const float* x       = (const float*)d_in[0];
  const float* y       = (const float*)d_in[1];
  const float* w_cq    = (const float*)d_in[2];
  const float* w_ckv   = (const float*)d_in[3];
  const float* ln_q_w  = (const float*)d_in[4];
  const float* ln_q_b  = (const float*)d_in[5];
  const float* ln_kv_w = (const float*)d_in[6];
  const float* ln_kv_b = (const float*)d_in[7];
  const float* w_kv    = (const float*)d_in[8];
  const float* w_kvdw  = (const float*)d_in[9];
  const float* w_q     = (const float*)d_in[10];
  const float* temp    = (const float*)d_in[11];
  const float* w_po    = (const float*)d_in[12];
  const float* w_exp   = (const float*)d_in[13];
  const float* ln_o_w  = (const float*)d_in[14];
  const float* ln_o_b  = (const float*)d_in[15];
  const float* w_ffn1  = (const float*)d_in[16];
  const float* w_ffdw  = (const float*)d_in[17];
  const float* w_ffn2  = (const float*)d_in[18];
  float* out = (float*)d_out;

  float* wsf = (float*)d_ws;
  const size_t SB = (size_t)BB * SCH * HWC;   // 8.39M floats (33.5 MB)
  float* buf0   = wsf;             // xq (LN fused) -> later attention output o
  float* buf1   = wsf + 1 * SB;    // ykv (LN fused) -> later project_out result
  float* bufq   = wsf + 2 * SB;    // q
  float* bufkv0 = wsf + 3 * SB;    // kv pre-depthwise  (2*SB)
  float* bufkv1 = wsf + 5 * SB;    // kv (k|v)          (2*SB)
  float* bufc0  = wsf + 7 * SB;    // vexp / vn in-place (4*SB)
  float* bufc1  = wsf + 11 * SB;   // ffn hidden         (4*SB)
  float* buffdw = bufkv0;          // reuse kv region (4*SB) for dw+gelu out
  float* rnq    = wsf + 15 * SB;
  float* rnk    = rnq + 512;
  float* attnb  = rnq + 1024;      // 4096 floats

  dim3 blk(256);
  dim3 gS (HWC / 256, 1, BB);      // cout = 64
  dim3 gKV(HWC / 256, 2, BB);      // cout = 128
  dim3 gC (HWC / 256, 4, BB);      // cout = 256
  dim3 gPx(HWC / 256, BB);

  // compress convs with fused channel-LayerNorm
  wmma_conv<<<gS, blk, 0, stream>>>(x, nullptr, CCH, CCH, w_cq,  buf0, SCH, 3,
                                    ln_q_w, ln_q_b, 1);
  wmma_conv<<<gS, blk, 0, stream>>>(y, nullptr, CCH, CCH, w_ckv, buf1, SCH, 3,
                                    ln_kv_w, ln_kv_b, 1);
  // kv = dw3x3(1x1(ykv)), q = 3x3(xq)
  wmma_conv<<<gKV, blk, 0, stream>>>(buf1, nullptr, SCH, SCH, w_kv, bufkv0,
                                     2 * SCH, 1, nullptr, nullptr, 0);
  dw_conv<<<dim3(HWC / 256, 2 * SCH, BB), blk, 0, stream>>>(bufkv0, w_kvdw, bufkv1, 0);
  wmma_conv<<<gS, blk, 0, stream>>>(buf0, nullptr, SCH, SCH, w_q, bufq, SCH, 3,
                                    nullptr, nullptr, 0);
  // channel attention (tiny 8x8 per head)
  rownorm_kernel<<<dim3(BB * SCH, 2), blk, 0, stream>>>(bufq, bufkv1, rnq, rnk);
  attn_mat<<<dim3(BB * NHEAD), blk, 0, stream>>>(bufq, bufkv1, rnq, rnk, temp, attnb);
  attn_apply<<<dim3(HWC / 256, BB * NHEAD), blk, 0, stream>>>(attnb, bufkv1, buf0);
  // project out, expand, LN (C=256, standalone)
  wmma_conv<<<gS, blk, 0, stream>>>(buf0, nullptr, SCH, SCH, w_po, buf1, SCH, 1,
                                    nullptr, nullptr, 0);
  wmma_conv<<<gC, blk, 0, stream>>>(buf1, nullptr, SCH, SCH, w_exp, bufc0, CCH, 3,
                                    nullptr, nullptr, 0);
  cln_kernel<<<gPx, blk, 0, stream>>>(bufc0, ln_o_w, ln_o_b, CCH);
  // FFN: 1x1 on concat(x, vn) -> dw3x3 + gelu -> 1x1 -> out
  wmma_conv<<<gC, blk, 0, stream>>>(x, bufc0, CCH, 2 * CCH, w_ffn1, bufc1, CCH, 1,
                                    nullptr, nullptr, 0);
  dw_conv<<<dim3(HWC / 256, CCH, BB), blk, 0, stream>>>(bufc1, w_ffdw, buffdw, 1);
  wmma_conv<<<gC, blk, 0, stream>>>(buffdw, nullptr, CCH, CCH, w_ffn2, out, CCH, 1,
                                    nullptr, nullptr, 0);
}
